// Encoder_57380763074767
// MI455X (gfx1250) — matrix operand order repaired and checked by reading, not compiled, not run on hardware
//
#include <hip/hip_runtime.h>

// ---------------- problem constants ----------------
#define HH   1024          // hidden size
#define BB   4096          // sequence length (scan axis)
#define GG   4096          // 4*H gate rows
#define KP   64            // padded input feature count (real = 33)
#define NWG  64            // persistent workgroups
#define TPB  256           // threads per block

typedef __attribute__((ext_vector_type(16))) __bf16 v16bf;
typedef __attribute__((ext_vector_type(8)))  float  v8f;
typedef __attribute__((ext_vector_type(8)))  __bf16 bf8v;
typedef __attribute__((ext_vector_type(4)))  __bf16 bf4v;

// ---------------- workspace layout (bytes) ----------------
#define OFF_X0    ((size_t)0)                              // BB*GG*4 = 64 MB
#define OFF_W0B   (OFF_X0   + (size_t)BB*GG*4)             // GG*KP*2
#define OFF_WHH0  (OFF_W0B  + (size_t)GG*KP*2)             // GG*HH*2
#define OFF_WIH1  (OFF_WHH0 + (size_t)GG*HH*2)
#define OFF_WHH1  (OFF_WIH1 + (size_t)GG*HH*2)
#define OFF_XINB  (OFF_WHH1 + (size_t)GG*HH*2)             // BB*KP*2
#define OFF_GT0   (OFF_XINB + (size_t)BB*KP*2)             // GG*4
#define OFF_GT1   (OFF_GT0  + (size_t)GG*4)
#define OFF_CNT   (OFF_GT1  + (size_t)GG*4)

__device__ __forceinline__ float sigf(float x){ return 1.0f/(1.0f + __expf(-x)); }
__device__ __forceinline__ float tanhfast(float x){
  float e = __expf(-2.0f*x);
  return (1.0f - e) / (1.0f + e);
}

// ---------------- prep: embeddings -> packed bf16 input rows ----------------
__global__ __launch_bounds__(TPB) void embed_pack_kernel(
    const float* __restrict__ x, const int* __restrict__ ip, const int* __restrict__ port,
    const float* __restrict__ ip_emb, const float* __restrict__ port_emb,
    __bf16* __restrict__ xinb)
{
  int t = blockIdx.x*blockDim.x + threadIdx.x;
  if (t >= BB) return;
  __bf16* row = xinb + (size_t)t*KP;
  #pragma unroll
  for (int k=0;k<17;++k) row[k] = (__bf16)x[t*17+k];
  #pragma unroll
  for (int j=0;j<8;++j)  row[17+j] = (__bf16)ip_emb[ip[t*8+j]];   // ip_emb is (256,1)
  #pragma unroll
  for (int j=0;j<2;++j){
    int p = port[t*2+j];
    #pragma unroll
    for (int d=0;d<4;++d) row[25 + j*4 + d] = (__bf16)port_emb[p*4+d];
  }
  #pragma unroll
  for (int k=33;k<KP;++k) row[k] = (__bf16)0.0f;
}

// ---------------- prep: W_ih0 (4096x33) -> bf16 padded (4096x64) ----------------
__global__ __launch_bounds__(TPB) void pad_w0_kernel(
    const float* __restrict__ w, __bf16* __restrict__ w0b)
{
  int g = blockIdx.x*blockDim.x + threadIdx.x;
  if (g >= GG) return;
  __bf16* row = w0b + (size_t)g*KP;
  #pragma unroll
  for (int k=0;k<33;++k) row[k] = (__bf16)w[g*33+k];
  #pragma unroll
  for (int k=33;k<KP;++k) row[k] = (__bf16)0.0f;
}

// ---------------- prep: fp32 -> bf16 (vectorized) ----------------
__global__ __launch_bounds__(TPB) void cvt_bf16_kernel(
    const float* __restrict__ src, __bf16* __restrict__ dst, int n4)
{
  int i = blockIdx.x*blockDim.x + threadIdx.x;
  if (i >= n4) return;
  float4 v = ((const float4*)src)[i];
  bf4v o;
  o.x = (__bf16)v.x; o.y = (__bf16)v.y; o.z = (__bf16)v.z; o.w = (__bf16)v.w;
  ((bf4v*)dst)[i] = o;
}

// ---------------- WMMA GEMM: X0[t][g] = xin[t]·W_ih0[g] + b_ih0[g] + b_hh0[g] ----
// One wave per 16x16 C tile, K = 64 (2 k-tiles of 32), bf16 WMMA.
// Fragment packing: A and B both use contiguous k = 16*half + i (16-bit A 16x32, B 32x16, f32 C 16x16).
__global__ __launch_bounds__(128) void gemm_x0_kernel(
    const __bf16* __restrict__ xinb, const __bf16* __restrict__ w0b,
    const float* __restrict__ bih0, const float* __restrict__ bhh0,
    float* __restrict__ X0)
{
  const int lane = threadIdx.x & 31;
  const int wave = threadIdx.x >> 5;
  const int tile = blockIdx.x*4 + wave;      // 65536 tiles
  const int tm = tile >> 8;                  // time tile   (0..255)
  const int tn = tile & 255;                 // gate tile   (0..255)
  const int M0 = tm*16, N0 = tn*16;
  const int half = lane >> 4;
  const int sub  = lane & 15;

  v8f c = {};
  #pragma unroll
  for (int kt=0; kt<2; ++kt){
    const __bf16* arow = xinb + (size_t)(M0+sub)*KP + kt*32;
    const __bf16* brow = w0b  + (size_t)(N0+sub)*KP + kt*32;
    v16bf a, b;
    #pragma unroll
    for (int i=0;i<16;++i){
      int koff = half*16 + i;                // A-matrix: K striped by half-wave
      a[i] = arow[koff];
      b[i] = brow[half*16 + i];              // B-matrix: K striped by half-wave
    }
    c = __builtin_amdgcn_wmma_f32_16x16x32_bf16(false, a, false, b,
                                                (short)0, c, false, false);
  }
  const int g = N0 + sub;
  const float bias = bih0[g] + bhh0[g];
  #pragma unroll
  for (int r=0;r<8;++r){
    int m = M0 + r + 8*half;                 // C layout: VGPR r -> M = r (+8 for upper half)
    X0[(size_t)m*GG + g] = c[r] + bias;
  }
}

// ---------------- device-wide barrier: monotonic counter ----------------
__device__ __forceinline__ void grid_barrier(unsigned int* cnt, unsigned int& gen){
  __syncthreads();
  if (threadIdx.x == 0){
    gen += 1;
    __threadfence();
    __hip_atomic_fetch_add(cnt, 1u, __ATOMIC_RELEASE, __HIP_MEMORY_SCOPE_AGENT);
    while (__hip_atomic_load(cnt, __ATOMIC_ACQUIRE, __HIP_MEMORY_SCOPE_AGENT)
           < gen * (unsigned)NWG){
      __builtin_amdgcn_s_sleep(2);
    }
    __threadfence();
  }
  __syncthreads();
}

// ---------------- persistent 2-layer LSTM scan ----------------
// 64 blocks x 256 threads. Each block: rows [b*64, b*64+64), 4 lanes/row, 256 K each.
// h/c state replicated per block in LDS; gates cross the device barrier via L2.
__global__ __launch_bounds__(TPB, 1) void lstm_persist_kernel(
    const float* __restrict__ X0,
    const __bf16* __restrict__ Whh0,
    const __bf16* __restrict__ Wih1,
    const __bf16* __restrict__ Whh1,
    float* __restrict__ gt0, float* __restrict__ gt1,
    unsigned int* __restrict__ cnt,
    const float* __restrict__ hidden, const float* __restrict__ cellv,
    const float* __restrict__ bih1, const float* __restrict__ bhh1,
    float* __restrict__ out)
{
  __shared__ float h0[HH], c0[HH], h1[HH], c1[HH];
  const int b   = blockIdx.x;
  const int tid = threadIdx.x;

  for (int j=tid; j<HH; j+=TPB){
    h0[j] = hidden[j];      h1[j] = hidden[HH+j];
    c0[j] = cellv[j];       c1[j] = cellv[HH+j];
  }
  __syncthreads();

  const int row   = b*64 + (tid>>2);
  const int sub   = tid & 3;
  const int kbase = sub*256;
  const __bf16* w0r = Whh0 + (size_t)row*HH + kbase;
  const __bf16* w1r = Wih1 + (size_t)row*HH + kbase;
  const __bf16* w2r = Whh1 + (size_t)row*HH + kbase;
  const float rbias1 = bih1[row] + bhh1[row];

  unsigned int gen = 0;

  for (int t=0; t<BB; ++t){
    // ---- layer 0 recurrent GEMV: gates0 = X0[t] + Whh0 @ h0 ----
    {
      const float4* h4 = (const float4*)(h0 + kbase);
      float s = 0.f;
      #pragma unroll 4
      for (int it=0; it<32; ++it){
        bf8v w = *(const bf8v*)(w0r + 8*it);
        float4 ha = h4[2*it], hb = h4[2*it+1];
        s += (float)w[0]*ha.x + (float)w[1]*ha.y + (float)w[2]*ha.z + (float)w[3]*ha.w
           + (float)w[4]*hb.x + (float)w[5]*hb.y + (float)w[6]*hb.z + (float)w[7]*hb.w;
      }
      s += __shfl_xor(s, 1, 4);
      s += __shfl_xor(s, 2, 4);
      if (sub == 0) gt0[row] = s + X0[(size_t)t*GG + row];
    }
    grid_barrier(cnt, gen);

    // ---- layer 0 cell update (redundant per block) ----
    for (int j=tid; j<HH; j+=TPB){
      float ig = sigf(gt0[j]);
      float fg = sigf(gt0[HH + j]);
      float gg = tanhfast(gt0[2*HH + j]);
      float og = sigf(gt0[3*HH + j]);
      float cn = fg*c0[j] + ig*gg;
      c0[j] = cn;
      h0[j] = og * tanhfast(cn);
    }
    __syncthreads();

    // ---- layer 1 GEMV: gates1 = Wih1 @ h0' + Whh1 @ h1 + biases ----
    {
      const float4* ha4 = (const float4*)(h0 + kbase);
      const float4* hb4 = (const float4*)(h1 + kbase);
      float s = 0.f;
      #pragma unroll 4
      for (int it=0; it<32; ++it){
        bf8v wa = *(const bf8v*)(w1r + 8*it);
        float4 a0 = ha4[2*it], a1 = ha4[2*it+1];
        s += (float)wa[0]*a0.x + (float)wa[1]*a0.y + (float)wa[2]*a0.z + (float)wa[3]*a0.w
           + (float)wa[4]*a1.x + (float)wa[5]*a1.y + (float)wa[6]*a1.z + (float)wa[7]*a1.w;
        bf8v wb = *(const bf8v*)(w2r + 8*it);
        float4 b0 = hb4[2*it], b1 = hb4[2*it+1];
        s += (float)wb[0]*b0.x + (float)wb[1]*b0.y + (float)wb[2]*b0.z + (float)wb[3]*b0.w
           + (float)wb[4]*b1.x + (float)wb[5]*b1.y + (float)wb[6]*b1.z + (float)wb[7]*b1.w;
      }
      s += __shfl_xor(s, 1, 4);
      s += __shfl_xor(s, 2, 4);
      if (sub == 0) gt1[row] = s + rbias1;
    }
    grid_barrier(cnt, gen);

    // ---- layer 1 cell update (redundant per block) ----
    for (int j=tid; j<HH; j+=TPB){
      float ig = sigf(gt1[j]);
      float fg = sigf(gt1[HH + j]);
      float gg = tanhfast(gt1[2*HH + j]);
      float og = sigf(gt1[3*HH + j]);
      float cn = fg*c1[j] + ig*gg;
      c1[j] = cn;
      h1[j] = og * tanhfast(cn);
    }
    __syncthreads();

    // ---- write outputs[t], block owns 16 columns ----
    if (tid < 16){
      int j = b*16 + tid;
      float v = h1[j];
      out[(size_t)t*HH + j] = v > 0.f ? v : 0.f;
    }
  }

  // ---- final hidden/cell state: (hf[2][H], cf[2][H]) appended after outputs ----
  if (b == 0){
    const size_t tail = (size_t)BB*HH;
    for (int j=tid; j<HH; j+=TPB){
      out[tail          + j] = h0[j];
      out[tail +   HH   + j] = h1[j];
      out[tail + 2*HH   + j] = c0[j];
      out[tail + 3*HH   + j] = c1[j];
    }
  }
}

// ---------------- launcher ----------------
extern "C" void kernel_launch(void* const* d_in, const int* in_sizes, int n_in,
                              void* d_out, int out_size, void* d_ws, size_t ws_size,
                              hipStream_t stream)
{
  const float* x        = (const float*)d_in[0];
  const int*   ip       = (const int*)  d_in[1];
  const int*   port     = (const int*)  d_in[2];
  const float* hidden   = (const float*)d_in[3];
  const float* cellv    = (const float*)d_in[4];
  const float* ip_emb   = (const float*)d_in[5];
  const float* port_emb = (const float*)d_in[6];
  const float* W_ih0    = (const float*)d_in[7];
  const float* W_hh0    = (const float*)d_in[8];
  const float* b_ih0    = (const float*)d_in[9];
  const float* b_hh0    = (const float*)d_in[10];
  const float* W_ih1    = (const float*)d_in[11];
  const float* W_hh1    = (const float*)d_in[12];
  const float* b_ih1    = (const float*)d_in[13];
  const float* b_hh1    = (const float*)d_in[14];
  float* out = (float*)d_out;

  char* ws = (char*)d_ws;
  float*        X0    = (float*)       (ws + OFF_X0);
  __bf16*       W0B   = (__bf16*)      (ws + OFF_W0B);
  __bf16*       WHH0  = (__bf16*)      (ws + OFF_WHH0);
  __bf16*       WIH1  = (__bf16*)      (ws + OFF_WIH1);
  __bf16*       WHH1  = (__bf16*)      (ws + OFF_WHH1);
  __bf16*       XINB  = (__bf16*)      (ws + OFF_XINB);
  float*        GT0   = (float*)       (ws + OFF_GT0);
  float*        GT1   = (float*)       (ws + OFF_GT1);
  unsigned int* CNT   = (unsigned int*)(ws + OFF_CNT);

  // prep: pack inputs + bf16 weight copies
  embed_pack_kernel<<<BB/TPB, TPB, 0, stream>>>(x, ip, port, ip_emb, port_emb, XINB);
  pad_w0_kernel    <<<GG/TPB, TPB, 0, stream>>>(W_ih0, W0B);

  const int n4 = GG*HH/4;                    // 1,048,576 float4 per matrix
  cvt_bf16_kernel<<<n4/TPB, TPB, 0, stream>>>(W_hh0, WHH0, n4);
  cvt_bf16_kernel<<<n4/TPB, TPB, 0, stream>>>(W_ih1, WIH1, n4);
  cvt_bf16_kernel<<<n4/TPB, TPB, 0, stream>>>(W_hh1, WHH1, n4);

  // time-parallel input projection via bf16 WMMA (65536 tiles, 4 waves/block)
  gemm_x0_kernel<<<(BB/16)*(GG/16)/4, 128, 0, stream>>>(XINB, W0B, b_ih0, b_hh0, X0);

  // reset device barrier counter, then run the persistent sequential scan
  hipMemsetAsync(CNT, 0, 64, stream);
  lstm_persist_kernel<<<NWG, TPB, 0, stream>>>(X0, WHH0, WIH1, WHH1, GT0, GT1, CNT,
                                               hidden, cellv, b_ih1, b_hh1, out);
}
